// DecoderDirected_65532611002917
// MI455X (gfx1250) — matrix operand order repaired and checked by reading, not compiled, not run on hardware
//
#include <hip/hip_runtime.h>
#include <hip/hip_bf16.h>
#include <math.h>

typedef __attribute__((ext_vector_type(16))) _Float16 v16h;
typedef __attribute__((ext_vector_type(8)))  float    v8f;

#define B_     8
#define NMAX_  128
#define PE_    64
#define E_     5
#define DS_    64
#define NF_    8
#define COLS_  1280          // 4*E*PE
#define TOTAL_ 1024          // B*NMAX

// workspace layout (float element offsets)
#define WS_ALPHA 0           // 8 floats
#define WS_U     16          // 64 floats
#define WS_BSV   80          // 64 floats
#define WS_PROJ  256         // 1024*1280 halves (= 655360 float slots)
#define WS_Z     655616      // 655360 floats
#define WS_S     1310976     // 5120 floats

// output layout (floats)
#define OUT_EHAT  0          // (8,128,128,5)  = 655360
#define OUT_XHAT  655360     // (1024,8)       = 8192
#define OUT_EMASK 663552     // (8,128,128)    = 131072
#define OUT_MASK  794624     // (8,128)        = 1024

static __device__ __forceinline__ unsigned pack_f16x2(float a, float b) {
  union { _Float16 h[2]; unsigned u; } r;
  r.h[0] = (_Float16)a;
  r.h[1] = (_Float16)b;
  return r.u;
}

// ---------------------------------------------------------------------------
// K0: alpha[b] = (2-cos(2*pi*q))/sin(2*pi*q);  u = (w_phi^T w_ctx)/127;
//     bsv = b_self + b_phi^T w_ctx
// ---------------------------------------------------------------------------
__global__ void k0_prep(const float* __restrict__ q, const float* __restrict__ w_phi,
                        const float* __restrict__ b_phi, const float* __restrict__ w_ctx,
                        const float* __restrict__ b_self, float* __restrict__ ws) {
  int t = threadIdx.x;                 // 64 threads
  if (t < B_) {
    float ang = 6.28318530718f * q[t];
    ws[WS_ALPHA + t] = (2.0f - cosf(ang)) / sinf(ang);
  }
  float su = 0.f, sv = 0.f;
  for (int d = 0; d < DS_; ++d) {
    float wc = w_ctx[d * DS_ + t];
    su = fmaf(w_phi[d], wc, su);
    sv = fmaf(b_phi[d], wc, sv);
  }
  ws[WS_U + t]   = su * (1.0f / 127.0f);
  ws[WS_BSV + t] = b_self[t] + sv;
}

// ---------------------------------------------------------------------------
// K1: proj = x @ w_attn + b_attn, stored fp16; k1 section (*scale),
//     k2 section (*scale*alpha[b]) pre-scaled so K2 is pure accumulate.
//     One wave per 16x16 tile, 2x v_wmma_f32_16x16x32_f16 over K=64.
//     A fragments loaded as float4 (K-contiguous WMMA layout).
// ---------------------------------------------------------------------------
__global__ void k1_proj(const float* __restrict__ x, const float* __restrict__ w_attn,
                        const float* __restrict__ b_attn, float* __restrict__ ws) {
  _Float16* proj = (_Float16*)(ws + WS_PROJ);
  const float* alpha = ws + WS_ALPHA;
  const int lane = threadIdx.x & 31;
  const int wave = blockIdx.x * (blockDim.x >> 5) + (threadIdx.x >> 5);  // 5120 waves
  const int tM = wave / (COLS_ / 16);   // 64 row tiles
  const int tN = wave % (COLS_ / 16);   // 80 col tiles
  const int half = lane >> 4;
  const int mr = lane & 15;
  const int rowA = tM * 16 + mr;
  const float k_scale = 0.05590169944f;  // 1/sqrt(E*PE)

  v8f acc = {};
  for (int ks = 0; ks < 2; ++ks) {
    union { unsigned u[8]; v16h h; } A, Bf;
    const int kbA = half * 8 + ks * 32;
    // A: lanes hold row rowA, K = kbA..kbA+7 (v0..3) and kbA+16..kbA+23 (v4..7)
    const float4 p0 = *(const float4*)(x + rowA * PE_ + kbA);
    const float4 p1 = *(const float4*)(x + rowA * PE_ + kbA + 4);
    const float4 p2 = *(const float4*)(x + rowA * PE_ + kbA + 16);
    const float4 p3 = *(const float4*)(x + rowA * PE_ + kbA + 20);
    A.u[0] = pack_f16x2(p0.x, p0.y);  A.u[1] = pack_f16x2(p0.z, p0.w);
    A.u[2] = pack_f16x2(p1.x, p1.y);  A.u[3] = pack_f16x2(p1.z, p1.w);
    A.u[4] = pack_f16x2(p2.x, p2.y);  A.u[5] = pack_f16x2(p2.z, p2.w);
    A.u[6] = pack_f16x2(p3.x, p3.y);  A.u[7] = pack_f16x2(p3.z, p3.w);
    // B: column colB, K = kbB..kbB+7 and kbB+16..kbB+23 (stride-COLS_ gather, unavoidable)
    const int colB = tN * 16 + mr;
    const int kbB = half * 8 + ks * 32;
#pragma unroll
    for (int v = 0; v < 8; ++v) {
      const int k = kbB + 2 * v + (v >= 4 ? 8 : 0);
      Bf.u[v] = pack_f16x2(w_attn[k * COLS_ + colB], w_attn[(k + 1) * COLS_ + colB]);
    }
    acc = __builtin_amdgcn_wmma_f32_16x16x32_f16(false, A.h, false, Bf.h,
                                                 (short)0, acc, false, false);
  }
  const int b = (tM * 16) >> 7;
  const float al = alpha[b];
  const int col = tN * 16 + mr;
  const float factor = (col < 320) ? k_scale
                     : ((col >= 640 && col < 960) ? k_scale * al : 1.0f);
  const float bias = b_attn[col];
#pragma unroll
  for (int j = 0; j < 8; ++j) {
    const int row = tM * 16 + j + 8 * half;
    proj[row * COLS_ + col] = (_Float16)((acc[j] + bias) * factor);
  }
}

// ---------------------------------------------------------------------------
// K2: z[b,e,n,m] = (s*k1)·q1 + (s*a*k2)·q2 per 16x16 tile; 4 WMMAs per wave.
//     Fragments are K-contiguous fp16 pairs in proj -> uint4 (b128) loads.
// ---------------------------------------------------------------------------
__global__ void k2_z(float* __restrict__ ws) {
  const uint4* proj128 = (const uint4*)(ws + WS_PROJ);
  float* z = ws + WS_Z;
  const int lane = threadIdx.x & 31;
  int wave = blockIdx.x * (blockDim.x >> 5) + (threadIdx.x >> 5);  // 2560 waves
  const int mt = wave & 7; wave >>= 3;
  const int nt = wave & 7; wave >>= 3;
  const int e = wave % E_;
  const int b = wave / E_;
  const int half = lane >> 4;
  const int mr = lane & 15;
  const int rowA = b * NMAX_ + nt * 16 + mr;   // k-rows  (z rows,  M)
  const int rowB = b * NMAX_ + mt * 16 + mr;   // q-rows  (z cols,  N)
  const int cK1 = e * PE_;
  const int cQ1 = 320 + e * PE_;
  const int cK2 = 640 + e * PE_;
  const int cQ2 = 960 + e * PE_;

  v8f acc = {};
  for (int ks = 0; ks < 2; ++ks) {
    union { uint4 q[2]; v16h h; } A1, A2, B1, B2;
    const int kbA = half * 8 + ks * 32;    // halves
    const int kbB = half * 8 + ks * 32;    // halves
    // dword->uint4 indices: (row*COLS + col + kb) halves / 8 halves per uint4
    const int a1 = (rowA * COLS_ + cK1 + kbA) >> 3;
    const int a2 = (rowA * COLS_ + cK2 + kbA) >> 3;
    const int b1 = (rowB * COLS_ + cQ1 + kbB) >> 3;
    const int b2 = (rowB * COLS_ + cQ2 + kbB) >> 3;
    A1.q[0] = proj128[a1];     A1.q[1] = proj128[a1 + 2];   // K +0..7, +16..23
    A2.q[0] = proj128[a2];     A2.q[1] = proj128[a2 + 2];
    B1.q[0] = proj128[b1];     B1.q[1] = proj128[b1 + 2];   // K +0..7, +16..23
    B2.q[0] = proj128[b2];     B2.q[1] = proj128[b2 + 2];
    acc = __builtin_amdgcn_wmma_f32_16x16x32_f16(false, A1.h, false, B1.h,
                                                 (short)0, acc, false, false);
    acc = __builtin_amdgcn_wmma_f32_16x16x32_f16(false, A2.h, false, B2.h,
                                                 (short)0, acc, false, false);
  }
  const int m = mt * 16 + mr;
  const long base = (((long)(b * E_ + e)) * NMAX_ + nt * 16) * NMAX_;
#pragma unroll
  for (int j = 0; j < 8; ++j) {
    const int n = j + 8 * half;
    z[base + (long)n * NMAX_ + m] = acc[j];
  }
}

// ---------------------------------------------------------------------------
// K3: S[b,e,n] = sum_m z[b,e,n,m] - z[b,e,n,n]   (one wave per row)
// ---------------------------------------------------------------------------
__global__ void k3_rowsum(float* __restrict__ ws) {
  const float* z = ws + WS_Z;
  float* S = ws + WS_S;
  const int lane = threadIdx.x & 31;
  const int r = blockIdx.x * (blockDim.x >> 5) + (threadIdx.x >> 5);  // 5120 rows
  const float* row = z + (long)r * NMAX_;
  float s = row[lane] + row[lane + 32] + row[lane + 64] + row[lane + 96];
  for (int off = 16; off; off >>= 1) s += __shfl_xor(s, off, 32);
  if (lane == 0) {
    const int n = r & 127;
    S[r] = s - row[n];
  }
}

// ---------------------------------------------------------------------------
// K4: the transcendental-bound pass.
//     e_hat[b,m,n,e] = sum_d gelu(z*w_self[d] + S*u[d] + bsv[d]) * w_out[d] + b_out
// ---------------------------------------------------------------------------
__global__ void k4_energy(const float* __restrict__ ws, const float* __restrict__ w_self,
                          const float* __restrict__ w_out, const float* __restrict__ b_out,
                          float* __restrict__ out) {
  __shared__ float sws[DS_], su[DS_], sbv[DS_], swo[DS_];
  const int t = threadIdx.x;
  if (t < DS_) {
    sws[t] = w_self[t];
    su[t]  = ws[WS_U + t];
    sbv[t] = ws[WS_BSV + t];
    swo[t] = w_out[t];
  }
  __syncthreads();
  const float* z = ws + WS_Z;
  const float* S = ws + WS_S;
  const long idx = (long)blockIdx.x * blockDim.x + t;   // 655360 threads
  const float zv = z[idx];
  const float Sv = S[idx >> 7];
  float acc = 0.f;
#pragma unroll 8
  for (int d = 0; d < DS_; ++d) {
    const float a = fmaf(zv, sws[d], fmaf(Sv, su[d], sbv[d]));
    const float g = 0.5f * a * (1.0f + erff(a * 0.70710678f));
    acc = fmaf(g, swo[d], acc);
  }
  const int m = (int)(idx & 127);
  const int n = (int)((idx >> 7) & 127);
  const int be = (int)(idx >> 14);
  const int e = be % E_;
  const int b = be / E_;
  out[OUT_EHAT + (((long)(b * NMAX_ + m)) * NMAX_ + n) * E_ + e] = acc + b_out[0];
}

// ---------------------------------------------------------------------------
// K5: x_hat = x @ w_node + b_node; edge_mask (1 except diag); mask (all 1)
// ---------------------------------------------------------------------------
__global__ void k5_rest(const float* __restrict__ x, const float* __restrict__ w_node,
                        const float* __restrict__ b_node, float* __restrict__ out) {
  const int tid = blockIdx.x * blockDim.x + threadIdx.x;  // 131072 threads
  const int m = tid & 127, n = (tid >> 7) & 127;
  out[OUT_EMASK + tid] = (n != m) ? 1.0f : 0.0f;
  if (tid < TOTAL_ * NF_) {
    const int row = tid >> 3, c = tid & 7;
    float s = b_node[c];
    for (int k = 0; k < PE_; ++k) s = fmaf(x[row * PE_ + k], w_node[k * NF_ + c], s);
    out[OUT_XHAT + tid] = s;
  }
  if (tid < TOTAL_) out[OUT_MASK + tid] = 1.0f;
}

// ---------------------------------------------------------------------------
extern "C" void kernel_launch(void* const* d_in, const int* in_sizes, int n_in,
                              void* d_out, int out_size, void* d_ws, size_t ws_size,
                              hipStream_t stream) {
  const float* x      = (const float*)d_in[0];
  // d_in[1] = batch (sorted, full graphs) -> not needed on device
  const float* q      = (const float*)d_in[2];
  const float* w_attn = (const float*)d_in[3];
  const float* b_attn = (const float*)d_in[4];
  const float* w_node = (const float*)d_in[5];
  const float* b_node = (const float*)d_in[6];
  const float* w_phi  = (const float*)d_in[7];
  const float* b_phi  = (const float*)d_in[8];
  const float* w_ctx  = (const float*)d_in[9];
  const float* w_self = (const float*)d_in[10];
  const float* b_self = (const float*)d_in[11];
  const float* w_out  = (const float*)d_in[12];
  const float* b_out  = (const float*)d_in[13];
  float* out = (float*)d_out;
  float* ws  = (float*)d_ws;

  hipLaunchKernelGGL(k0_prep,   dim3(1),    dim3(64),  0, stream, q, w_phi, b_phi, w_ctx, b_self, ws);
  hipLaunchKernelGGL(k1_proj,   dim3(640),  dim3(256), 0, stream, x, w_attn, b_attn, ws);
  hipLaunchKernelGGL(k2_z,      dim3(320),  dim3(256), 0, stream, ws);
  hipLaunchKernelGGL(k3_rowsum, dim3(640),  dim3(256), 0, stream, ws);
  hipLaunchKernelGGL(k4_energy, dim3(2560), dim3(256), 0, stream, ws, w_self, w_out, b_out, out);
  hipLaunchKernelGGL(k5_rest,   dim3(512),  dim3(256), 0, stream, x, w_node, b_node, out);
}
